// AttentionHead_46084999086295
// MI455X (gfx1250) — matrix operand order repaired and checked by reading, not compiled, not run on hardware
//
#include <hip/hip_runtime.h>

#define N_CTX   4096
#define D_MODEL 2048

typedef __attribute__((ext_vector_type(16))) __bf16          v16bf;
typedef __attribute__((ext_vector_type(8)))  float           v8f;
typedef __attribute__((ext_vector_type(16))) unsigned short  v16u;
typedef __attribute__((ext_vector_type(8)))  unsigned short  v8u;
typedef __attribute__((ext_vector_type(4)))  unsigned short  v4u;
typedef __attribute__((ext_vector_type(4)))  float           v4f;
typedef __attribute__((ext_vector_type(4)))  unsigned int    v4ui;
typedef __attribute__((ext_vector_type(8)))  int             v8i;
typedef __attribute__((ext_vector_type(4)))  int             v4i;

typedef __attribute__((address_space(3))) unsigned short lds_u16;

#ifndef USE_TDM
#define USE_TDM 1
#endif

__device__ __forceinline__ unsigned short f32_to_bf16(float f) {
  unsigned int u = __builtin_bit_cast(unsigned int, f);
  u += 0x7FFFu + ((u >> 16) & 1u);   // round-to-nearest-even
  return (unsigned short)(u >> 16);
}

// Build a 16-half WMMA fragment from two 16-byte chunks (global or LDS).
__device__ __forceinline__ v16bf make_frag(const unsigned short* lo,
                                           const unsigned short* hi) {
  v8u a = *(const v8u*)lo;
  v8u b = *(const v8u*)hi;
  v16u r;
#pragma unroll
  for (int i = 0; i < 8; ++i) { r[i] = a[i]; r[i + 8] = b[i]; }
  return __builtin_bit_cast(v16bf, r);
}

__device__ __forceinline__ v8f wmma_bf16(v16bf a, v16bf b, v8f c) {
  return __builtin_amdgcn_wmma_f32_16x16x32_bf16(false, a, false, b,
                                                 (short)0, c, false, false);
}

__device__ __forceinline__ v8f v8f_zero() {
  v8f z = {0.f, 0.f, 0.f, 0.f, 0.f, 0.f, 0.f, 0.f};
  return z;
}

// ---------------------------------------------------------------- convert
__global__ void cvt_f32_bf16(const float* __restrict__ in,
                             unsigned short* __restrict__ out, int n4) {
  int i = blockIdx.x * blockDim.x + threadIdx.x;
  int stride = gridDim.x * blockDim.x;
  for (; i < n4; i += stride) {
    v4f v = *(const v4f*)(in + 4 * (size_t)i);
    v4u o;
#pragma unroll
    for (int c = 0; c < 4; ++c) o[c] = f32_to_bf16(v[c]);
    *(v4u*)(out + 4 * (size_t)i) = o;
  }
}

// ---------------------------------------------------------------- GEMM
// C[M,N] = A[M,K] * B[K,N], bf16 inputs, f32 accumulate.
// 256 threads = 8 waves; workgroup tile 32(M) x 256(N); wave tile 16 x 64.
template <bool OUT_BF16>
__global__ __launch_bounds__(256) void gemm_bf16_k(
    const unsigned short* __restrict__ A,
    const unsigned short* __restrict__ B,
    void* __restrict__ C, int M, int N, int K) {
  __shared__ unsigned short BT[256 * 32];   // B tile, transposed: [n][k]

  const int tid  = threadIdx.x;
  const int wave = tid >> 5, lane = tid & 31;
  const int m0 = blockIdx.y * 32, n0 = blockIdx.x * 256;
  const int mw = m0 + (wave & 1) * 16;      // wave's 16 rows
  const int nw = (wave >> 1) * 64;          // wave's 64 cols within tile
  const int nl    = lane & 15;
  const int koffA = (lane >> 4) << 3;       // C rows: 0 or 8
  const int kbB   = (lane >> 4) << 4;       // B frag: 0 or 16

  v8f acc[4];
#pragma unroll
  for (int t = 0; t < 4; ++t) acc[t] = v8f_zero();

  const unsigned short* Arow = A + (size_t)(mw + nl) * K;
  const int kk = tid & 31;                  // staging: k within tile
  const int nb = (tid >> 5) * 32;           // staging: col chunk

  for (int k0 = 0; k0 < K; k0 += 32) {
    __syncthreads();
    {   // stage B[k0..k0+31][n0..n0+255] transposed into LDS
      const unsigned short* bsrc = B + (size_t)(k0 + kk) * N + n0 + nb;
#pragma unroll
      for (int c = 0; c < 32; c += 8) {
        v8u v = *(const v8u*)(bsrc + c);
#pragma unroll
        for (int i = 0; i < 8; ++i) BT[(nb + c + i) * 32 + kk] = v[i];
      }
    }
    __syncthreads();
    v16bf a = make_frag(Arow + k0 + kbB, Arow + k0 + kbB + 8);
#pragma unroll
    for (int t = 0; t < 4; ++t) {
      const unsigned short* bp = &BT[(nw + t * 16 + nl) * 32 + kbB];
      v16bf b = make_frag(bp, bp + 8);
      acc[t] = wmma_bf16(a, b, acc[t]);
    }
  }

#pragma unroll
  for (int t = 0; t < 4; ++t) {
#pragma unroll
    for (int r = 0; r < 8; ++r) {
      const int row = mw + r + koffA;             // C layout: M = r + 8*(lane>=16)
      const int col = n0 + nw + t * 16 + nl;
      if (OUT_BF16)
        ((unsigned short*)C)[(size_t)row * N + col] = f32_to_bf16(acc[t][r]);
      else
        ((float*)C)[(size_t)row * N + col] = acc[t][r];
    }
  }
}

// ---------------------------------------------------------------- flash attn
// U[16,D] = softmax(mask(P_blk @ E^T)) @ E   (streamed over key blocks of 32)
__global__ __launch_bounds__(256) void flash_attn_k(
    const unsigned short* __restrict__ Pbf,   // [N_CTX, D]
    const unsigned short* __restrict__ Ebf,   // [N_CTX, D]
    unsigned short* __restrict__ Ubf) {       // [N_CTX, D]
  constexpr int D = D_MODEL;
  extern __shared__ char smem[];
  unsigned short* PblkL = (unsigned short*)smem;                      // [16][D] row-major (64KB)
  unsigned short* EjT   = (unsigned short*)(smem + 65536);            // [D][32] E block transposed (128KB)
  float*          Spart = (float*)(smem + 65536 + 131072);            // 8 waves * 256 (8KB)
  float*          Sfull = (float*)(smem + 65536 + 131072 + 8192);     // 16x32 (2KB)
  unsigned short* pe    = (unsigned short*)(smem + 65536 + 131072 + 8192 + 2048); // 16x32 bf16
  float*          alphaR = (float*)(smem + 65536 + 131072 + 8192 + 2048 + 1024);
  float*          mrow   = alphaR + 16;
  float*          lrow   = mrow + 16;

  const int tid  = threadIdx.x;
  const int wave = tid >> 5, lane = tid & 31;
  const int nl    = lane & 15;
  const int koffA = (lane >> 4) << 3;
  const int kbB   = (lane >> 4) << 4;
  const int row0  = blockIdx.x * 16;
  const int nbase = wave * 256;             // wave's 256-wide D slice of U
  const int jmax  = (row0 + 15) >> 5;       // causal: skip fully-masked key blocks

  // ---- stage the loop-invariant P row block [16][D] into LDS once ----
#if USE_TDM && __has_builtin(__builtin_amdgcn_tensor_load_to_lds)
  if (tid < 32) {
    const unsigned long long ga = (unsigned long long)(Pbf + (size_t)row0 * D);
    const unsigned lds_addr = (unsigned)(size_t)(lds_u16*)PblkL;
    // D# group 0: count=1, lds_addr, global_addr, type=2 (ISA 8.3)
    v4ui g0;
    g0[0] = 1u;
    g0[1] = lds_addr;
    g0[2] = (unsigned)(ga & 0xFFFFFFFFu);
    g0[3] = (unsigned)((ga >> 32) & 0x01FFFFFFu) | (2u << 30);
    // D# group 1 (ISA 8.4): data_size=1 (2B), tensor_dim0=2048, tensor_dim1=4096,
    // tile_dim0=2048, tile_dim1=16, tensor_dim0_stride=2048
    v8i g1;
    g1[0] = 0x00010000;                       // wg_mask=0 | data_size=1<<16
    g1[1] = (int)((unsigned)D_MODEL << 16);   // tensor_dim0[15:0] << 16
    g1[2] = (int)((unsigned)N_CTX << 16);     // tensor_dim0[31:16]=0 | tensor_dim1[15:0]<<16
    g1[3] = (int)((unsigned)D_MODEL << 16);   // tensor_dim1[31:16]=0 | tile_dim0<<16
    g1[4] = 16;                               // tile_dim1=16 | tile_dim2=0
    g1[5] = D_MODEL;                          // tensor_dim0_stride[31:0]
    g1[6] = 0;
    g1[7] = 0;
    v4i g2 = {0, 0, 0, 0};
    v4i g3 = {0, 0, 0, 0};
#if __clang_major__ >= 23
    v8i g4 = {0, 0, 0, 0, 0, 0, 0, 0};
    __builtin_amdgcn_tensor_load_to_lds(g0, g1, g2, g3, g4, 0);
#else
    __builtin_amdgcn_tensor_load_to_lds(g0, g1, g2, g3, 0);
#endif
    __builtin_amdgcn_s_wait_tensorcnt(0);
  }
#else
  {   // fallback: cooperative copy (contiguous, 256B per thread)
    const unsigned short* src = Pbf + (size_t)row0 * D + tid * 128;
    unsigned short* dst = PblkL + tid * 128;
#pragma unroll
    for (int c = 0; c < 128; c += 8) *(v8u*)(dst + c) = *(const v8u*)(src + c);
  }
#endif

  if (tid < 16) { mrow[tid] = -__builtin_inff(); lrow[tid] = 0.0f; }

  v8f acc[16];
#pragma unroll
  for (int t = 0; t < 16; ++t) acc[t] = v8f_zero();

  const int sk = tid & 31;                  // staging: key row
  const int sd = (tid >> 5) * 256;          // staging: d chunk
  const unsigned short* ProwL = PblkL + (size_t)nl * D;   // LDS, loop-invariant

  for (int j = 0; j <= jmax; ++j) {
    const int j32 = j << 5;
    __syncthreads();   // prev phase-B done with EjT/pe/alphaR; Pblk visible
    {   // stage E[j32..j32+31][:] transposed into LDS: EjT[d][k]
      const unsigned short* src = Ebf + (size_t)(j32 + sk) * D + sd;
      if (j < jmax)
        __builtin_prefetch(Ebf + (size_t)(j32 + 32 + sk) * D + sd, 0, 1);
#pragma unroll 4
      for (int c = 0; c < 256; c += 8) {
        v8u v = *(const v8u*)(src + c);
#pragma unroll
        for (int i = 0; i < 8; ++i) EjT[(size_t)(sd + c + i) * 32 + sk] = v[i];
      }
    }
    {   // phase A: partial S = P_blk @ E_j^T ; waves split 2 key tiles x 4 K-slices
      const int tile = wave >> 2;
      const int kbeg = (wave & 3) * 512;
      const unsigned short* Erow = Ebf + (size_t)(j32 + tile * 16 + nl) * D;
      v8f s = v8f_zero();
#pragma unroll 4
      for (int kk2 = 0; kk2 < 512; kk2 += 32) {
        const int k0 = kbeg + kk2;
        v16bf a = make_frag(ProwL + k0 + kbB, ProwL + k0 + kbB + 8);
        v16bf b = make_frag(Erow + k0 + kbB,  Erow + k0 + kbB + 8);
        s = wmma_bf16(a, b, s);
      }
      float* sp = Spart + wave * 256;
#pragma unroll
      for (int r = 0; r < 8; ++r) sp[(r + koffA) * 16 + nl] = s[r];
    }
    __syncthreads();
    // reduce 4 K-partials + causal mask -> Sfull[16][32]
    for (int e = tid; e < 512; e += 256) {
      const int mm = e >> 5, nn = e & 31;
      const int tl = nn >> 4, nc = nn & 15;
      float v = Spart[(tl * 4 + 0) * 256 + mm * 16 + nc]
              + Spart[(tl * 4 + 1) * 256 + mm * 16 + nc]
              + Spart[(tl * 4 + 2) * 256 + mm * 16 + nc]
              + Spart[(tl * 4 + 3) * 256 + mm * 16 + nc];
      Sfull[e] = ((j32 + nn) > (row0 + mm)) ? -__builtin_inff() : v;
    }
    __syncthreads();
    // online softmax row update (one thread per query row)
    if (tid < 16) {
      const float NEGINF = -__builtin_inff();
      float mo = mrow[tid], lo = lrow[tid];
      float rmax = NEGINF;
      for (int nn = 0; nn < 32; ++nn) rmax = fmaxf(rmax, Sfull[tid * 32 + nn]);
      float mn = fmaxf(mo, rmax);
      float al = (mn == NEGINF) ? 1.0f : __expf(mo - mn);
      float ssum = 0.0f;
      for (int nn = 0; nn < 32; ++nn) {
        float sv = Sfull[tid * 32 + nn];
        float p  = (sv == NEGINF) ? 0.0f : __expf(sv - mn);
        pe[tid * 32 + nn] = f32_to_bf16(p);
        ssum += p;
      }
      mrow[tid] = mn; lrow[tid] = lo * al + ssum; alphaR[tid] = al;
    }
    __syncthreads();
    {   // phase B: rescale U, U += Pexp @ E_blk (K=32 keys)
      v16bf a = make_frag(pe + nl * 32 + kbB, pe + nl * 32 + kbB + 8);
      float al[8];
#pragma unroll
      for (int r = 0; r < 8; ++r) al[r] = alphaR[r + koffA];
#pragma unroll
      for (int t = 0; t < 16; ++t) {
#pragma unroll
        for (int r = 0; r < 8; ++r) acc[t][r] *= al[r];
        const unsigned short* bp = EjT + (size_t)(nbase + t * 16 + nl) * 32 + kbB;
        v16bf b = make_frag(bp, bp + 8);
        acc[t] = wmma_bf16(a, b, acc[t]);
      }
    }
  }
  __syncthreads();
  // finalize: U /= l, store bf16
  float li[8];
#pragma unroll
  for (int r = 0; r < 8; ++r) li[r] = 1.0f / lrow[r + koffA];
#pragma unroll
  for (int t = 0; t < 16; ++t) {
#pragma unroll
    for (int r = 0; r < 8; ++r) {
      Ubf[(size_t)(row0 + r + koffA) * D + nbase + t * 16 + nl] =
          f32_to_bf16(acc[t][r] * li[r]);
    }
  }
}

// ---------------------------------------------------------------- launch
extern "C" void kernel_launch(void* const* d_in, const int* in_sizes, int n_in,
                              void* d_out, int out_size, void* d_ws, size_t ws_size,
                              hipStream_t stream) {
  (void)in_sizes; (void)n_in; (void)out_size; (void)ws_size;
  const float* emb = (const float*)d_in[0];
  const float* qk  = (const float*)d_in[1];
  const float* ov  = (const float*)d_in[2];
  float* out = (float*)d_out;

  char* ws = (char*)d_ws;
  unsigned short* Ebf  = (unsigned short*)(ws + 0);          // 16 MB
  unsigned short* QKbf = (unsigned short*)(ws + 16777216);   //  8 MB
  unsigned short* OVbf = (unsigned short*)(ws + 25165824);   //  8 MB
  unsigned short* Pbf  = (unsigned short*)(ws + 33554432);   // 16 MB
  unsigned short* Ubf  = (unsigned short*)(ws + 50331648);   // 16 MB (total 64 MB)

  const int nE = N_CTX * D_MODEL;
  const int nW = D_MODEL * D_MODEL;

  cvt_f32_bf16<<<dim3(512), dim3(256), 0, stream>>>(emb, Ebf, nE / 4);
  cvt_f32_bf16<<<dim3(512), dim3(256), 0, stream>>>(qk,  QKbf, nW / 4);
  cvt_f32_bf16<<<dim3(512), dim3(256), 0, stream>>>(ov,  OVbf, nW / 4);

  // P = E @ qk   (bf16 out)
  gemm_bf16_k<true><<<dim3(D_MODEL / 256, N_CTX / 32), dim3(256), 0, stream>>>(
      Ebf, QKbf, (void*)Pbf, N_CTX, D_MODEL, D_MODEL);

  // U = softmax(mask(P @ E^T)) @ E  (~203 KB dynamic LDS; CDNA5 WGP has 320 KB)
  flash_attn_k<<<dim3(N_CTX / 16), dim3(256), 208064, stream>>>(Pbf, Ebf, Ubf);

  // out = U @ ov (fp32 out)
  gemm_bf16_k<false><<<dim3(D_MODEL / 256, N_CTX / 32), dim3(256), 0, stream>>>(
      Ubf, OVbf, (void*)out, N_CTX, D_MODEL, D_MODEL);
}
